// OnnxAttention_7662221656204
// MI455X (gfx1250) — compile-verified, matrix operand order checked by reading, not run on hardware
//
#include <hip/hip_runtime.h>
#include <hip/hip_bf16.h>

// ---------------- problem constants ----------------
#define Bsz   1
#define Tq    2048
#define Cdim  2048
#define Hh    16
#define KVh   4
#define Dd    128
#define PASTn 2048
#define Sall  (PASTn + Tq)        // 4096
#define Gg    (Hh / KVh)          // 4
#define NQKV  3072                // H*D + 2*KV*D

typedef __bf16 bf16;
typedef bf16  bf16x8  __attribute__((ext_vector_type(8)));
typedef bf16  bf16x16 __attribute__((ext_vector_type(16)));
typedef float f32x8   __attribute__((ext_vector_type(8)));

// round-to-nearest-even f32 -> bf16
__device__ __forceinline__ bf16 to_bf16(float f) {
  union { float f; unsigned u; } a; a.f = f;
  unsigned r = a.u + 0x7FFFu + ((a.u >> 16) & 1u);
  unsigned short h = (unsigned short)(r >> 16);
  union { unsigned short s; bf16 b; } o; o.s = h; return o.b;
}

// ---- CDNA5 async global->LDS copy (ASYNCcnt path, 16B per lane) ----
__device__ __forceinline__ void async_ld_b128(unsigned lds_byte_addr,
                                              const void* gaddr) {
  asm volatile("global_load_async_to_lds_b128 %0, %1, off"
               :: "v"(lds_byte_addr), "v"(gaddr) : "memory");
}
template <int N>
__device__ __forceinline__ void wait_async() {
#if __has_builtin(__builtin_amdgcn_s_wait_asynccnt)
  __builtin_amdgcn_s_wait_asynccnt(N);
#else
  asm volatile("s_wait_asynccnt %0" :: "i"(N) : "memory");
#endif
}

// Load one 16x32 bf16 WMMA operand fragment from a k-contiguous tile.
// Lane L: row = base_row + (L&15); lanes 0-15 take K {0..7,16..23}+k0,
// lanes 16-31 take K {8..15,24..31}+k0 -> two 16-byte contiguous loads.
__device__ __forceinline__ bf16x16 ld_frag(const bf16* tile, int base_row,
                                           int stride, int k0, int lane) {
  const bf16* p = tile + (size_t)(base_row + (lane & 15)) * stride
                       + k0 + ((lane >> 4) << 3);
  bf16x8 lo = *(const bf16x8*)(p);
  bf16x8 hi = *(const bf16x8*)(p + 16);
  bf16x16 r;
#pragma unroll
  for (int i = 0; i < 8; ++i) { r[i] = lo[i]; r[i + 8] = hi[i]; }
  return r;
}

// ---------------- cast kernels ----------------
__global__ void cast_f32_bf16(const float* __restrict__ src, bf16* __restrict__ dst, int n) {
  int i = blockIdx.x * 256 + threadIdx.x;
  if (i < n) dst[i] = to_bf16(src[i]);
}

// transpose-cast: src [rows][cols] f32 -> dst[(rowoff + c)*ldt + r] bf16
__global__ void cast_pack_T(const float* __restrict__ src, bf16* __restrict__ dst,
                            int rows, int cols, int ldt, int rowoff) {
  int i = blockIdx.x * 256 + threadIdx.x;
  if (i < rows * cols) {
    int r = i / cols, c = i % cols;
    dst[(size_t)(rowoff + c) * ldt + r] = to_bf16(src[i]);
  }
}

// ------- tiled bf16 WMMA GEMM: C(f32,MxN) = A(bf16,[M][K]) * Bt(bf16,[N][K])^T
// Block = 256 threads = 8 waves, tile 128x128, per-wave 64x32.
// Both operand tiles are k-contiguous -> all staging through async-to-LDS,
// double-buffered: prefetch tile kt+1 while computing tile kt.
__global__ __launch_bounds__(256) void gemm_bf16t(
    const bf16* __restrict__ A, const bf16* __restrict__ Bt, float* __restrict__ C,
    int M, int N, int K, int lda, int ldc) {
  __shared__ __align__(16) bf16 As[2][128 * 32];
  __shared__ __align__(16) bf16 Bs[2][128 * 32];
  const int tid = threadIdx.x, lane = tid & 31, w = tid >> 5;
  const int m0 = blockIdx.y * 128, n0 = blockIdx.x * 128;
  const int wm = (w >> 2) * 64;
  const int wn = (w & 3) * 32;
  const unsigned as_u32 = (unsigned)(uintptr_t)&As[0][0];
  const unsigned bs_u32 = (unsigned)(uintptr_t)&Bs[0][0];

  f32x8 acc[4][2];
#pragma unroll
  for (int mi = 0; mi < 4; ++mi)
#pragma unroll
    for (int ni = 0; ni < 2; ++ni)
#pragma unroll
      for (int e = 0; e < 8; ++e) acc[mi][ni][e] = 0.0f;

  // issue one 128x32 A tile + 128x32 Bt tile: 2x512 chunks of 16B, 4/thread
  auto issue_tile = [&](int buf, int k0) {
    unsigned la = as_u32 + (unsigned)buf * 8192u;
    unsigned lb = bs_u32 + (unsigned)buf * 8192u;
#pragma unroll
    for (int i = 0; i < 2; ++i) {
      int c = tid + i * 256;
      async_ld_b128(la + ((unsigned)c << 4),
                    A  + (size_t)(m0 + (c >> 2)) * lda + k0 + ((c & 3) << 3));
      async_ld_b128(lb + ((unsigned)c << 4),
                    Bt + (size_t)(n0 + (c >> 2)) * K   + k0 + ((c & 3) << 3));
    }
  };

  const int nk = K / 32;
  issue_tile(0, 0);
  for (int kt = 0; kt < nk; ++kt) {
    if (kt + 1 < nk) {
      issue_tile((kt + 1) & 1, (kt + 1) * 32);
      wait_async<4>();          // current tile's 4 chunks done; next in flight
    } else {
      wait_async<0>();
    }
    if (kt + 2 < nk) {          // L2 warm-up for the tile after next
      __builtin_prefetch(A  + (size_t)(m0 + (tid >> 1)) * lda + (kt + 2) * 32, 0, 1);
      __builtin_prefetch(Bt + (size_t)(n0 + (tid >> 1)) * K   + (kt + 2) * 32, 0, 1);
    }
    __syncthreads();

    const bf16* Asb = &As[kt & 1][0];
    const bf16* Bsb = &Bs[kt & 1][0];
    bf16x16 af[4], bfm[2];
#pragma unroll
    for (int mi = 0; mi < 4; ++mi) af[mi]  = ld_frag(Asb, wm + mi * 16, 32, 0, lane);
#pragma unroll
    for (int ni = 0; ni < 2; ++ni) bfm[ni] = ld_frag(Bsb, wn + ni * 16, 32, 0, lane);
#pragma unroll
    for (int mi = 0; mi < 4; ++mi)
#pragma unroll
      for (int ni = 0; ni < 2; ++ni)
        acc[mi][ni] = __builtin_amdgcn_wmma_f32_16x16x32_bf16(
            false, af[mi], false, bfm[ni], (short)0, acc[mi][ni], false, false);
    __syncthreads();
  }

#pragma unroll
  for (int mi = 0; mi < 4; ++mi)
#pragma unroll
    for (int ni = 0; ni < 2; ++ni)
#pragma unroll
      for (int v = 0; v < 8; ++v) {
        int row = m0 + wm + mi * 16 + v + ((lane >> 4) << 3);
        int col = n0 + wn + ni * 16 + (lane & 15);
        C[(size_t)row * ldc + col] = acc[mi][ni][v];
      }
}

// ---------------- RoPE + RMSNorm for Q -> bf16 [t][h][d] ----------------
__global__ __launch_bounds__(128) void postproc_q(
    const float* __restrict__ qkv, const float* __restrict__ cosp,
    const float* __restrict__ sinp, bf16* __restrict__ qb) {
  int b = blockIdx.x;            // t*H + h
  int t = b >> 4, h = b & 15;
  int d = threadIdx.x, j = d & 63;
  const float* row = qkv + (size_t)t * NQKV + h * Dd;
  float x1 = row[j], x2 = row[64 + j];
  float c = cosp[t * 64 + j], s = sinp[t * 64 + j];
  float r = (d < 64) ? (x1 * c + x2 * s) : (x2 * c - x1 * s);
  __shared__ float red[128];
  red[d] = r * r; __syncthreads();
  for (int off = 64; off > 0; off >>= 1) {
    if (d < off) red[d] += red[d + off];
    __syncthreads();
  }
  float out = r * rsqrtf(red[0] * (1.0f / 128.0f) + 1e-6f) * 1.2f;
  qb[(size_t)b * 128 + d] = to_bf16(out);
}

// ------- RoPE+RMSNorm for K, gate+ve for V; write cache outputs -------
// V also mirrored transposed: Vt[kv][d][s] for async-friendly attention B tiles.
__global__ __launch_bounds__(128) void postproc_kv(
    const float* __restrict__ qkv, const float* __restrict__ cosp,
    const float* __restrict__ sinp, const float* __restrict__ x,
    const float* __restrict__ ve, const float* __restrict__ Wg,
    float* __restrict__ outk, float* __restrict__ outv,
    bf16* __restrict__ Kall, bf16* __restrict__ Vt) {
  int b = blockIdx.x;            // t*KV + kv
  int t = b >> 2, kv = b & 3;
  int d = threadIdx.x, j = d & 63;
  __shared__ float red[128];
  __shared__ float gsh;
  const float* krow = qkv + (size_t)t * NQKV + Hh * Dd + kv * Dd;
  float x1 = krow[j], x2 = krow[64 + j];
  float c = cosp[t * 64 + j], s = sinp[t * 64 + j];
  float r = (d < 64) ? (x1 * c + x2 * s) : (x2 * c - x1 * s);
  red[d] = r * r;
  if (d == 0) {
    float g = 0.0f;
#pragma unroll
    for (int i = 0; i < 12; ++i) g += x[(size_t)t * Cdim + i] * Wg[i * KVh + kv];
    gsh = 3.0f / (1.0f + __expf(-g));
  }
  __syncthreads();
  for (int off = 64; off > 0; off >>= 1) {
    if (d < off) red[d] += red[d + off];
    __syncthreads();
  }
  float kk = r * rsqrtf(red[0] * (1.0f / 128.0f) + 1e-6f) * 1.2f;
  size_t oidx = ((size_t)(PASTn + t) * KVh + kv) * Dd + d;
  outk[oidx] = kk;
  Kall[oidx] = to_bf16(kk);
  float vv = qkv[(size_t)t * NQKV + Hh * Dd + KVh * Dd + kv * Dd + d]
           + gsh * ve[(size_t)t * (KVh * Dd) + kv * Dd + d];
  outv[oidx] = vv;
  Vt[(size_t)(kv * Dd + d) * Sall + (PASTn + t)] = to_bf16(vv);
}

// --------- copy past KV cache into outputs + bf16 mirrors (V transposed) ---------
__global__ void copy_past(const float* __restrict__ pk, const float* __restrict__ pv,
                          float* __restrict__ outk, float* __restrict__ outv,
                          bf16* __restrict__ Kall, bf16* __restrict__ Vt, int n) {
  int i = blockIdx.x * 256 + threadIdx.x;
  if (i < n) {
    float k = pk[i], v = pv[i];
    outk[i] = k; outv[i] = v;
    Kall[i] = to_bf16(k);
    int d = i & 127, kv = (i >> 7) & 3, s = i >> 9;
    Vt[(size_t)(kv * Dd + d) * Sall + s] = to_bf16(v);
  }
}

// ---------------- flash attention (bf16 WMMA, online softmax) ----------------
// grid = (T/128, H); block = 256 (8 waves); wave w owns 16 query rows.
// K [s][d] and Vt [d][s] chunks double-buffered through async-to-LDS.
__global__ __launch_bounds__(256) void flash_attn(
    const bf16* __restrict__ Qb,   // [T][H][D]
    const bf16* __restrict__ Kall, // [S][KV][D]
    const bf16* __restrict__ Vt,   // [KV][D][S]
    bf16* __restrict__ Ob) {       // [T][H*D]
  __shared__ __align__(16) bf16 Ks[2][64 * 128];  // [s][d]
  __shared__ __align__(16) bf16 Vs[2][128 * 64];  // [d][s]
  __shared__ __align__(16) bf16 Ps[8 * 16 * 64];  // per-wave P tile [m][s]
  const int tid = threadIdx.x, lane = tid & 31, w = tid >> 5;
  const int qt = blockIdx.x, h = blockIdx.y, kvh = h >> 2;
  const int q0 = qt * 128 + w * 16;
  const unsigned ks_u32 = (unsigned)(uintptr_t)&Ks[0][0];
  const unsigned vs_u32 = (unsigned)(uintptr_t)&Vs[0][0];

  // Q fragments: 16 rows x 128 (4 k-steps of 32), loaded straight from global
  bf16x16 qa[4];
#pragma unroll
  for (int c = 0; c < 4; ++c)
    qa[c] = ld_frag(Qb + h * Dd, q0, Hh * Dd, c * 32, lane);

  f32x8 o[8];
  float mi[8], li[8];
#pragma unroll
  for (int nt = 0; nt < 8; ++nt)
#pragma unroll
    for (int e = 0; e < 8; ++e) o[nt][e] = 0.0f;
#pragma unroll
  for (int v = 0; v < 8; ++v) { mi[v] = -1e30f; li[v] = 0.0f; }

  const int nch = (PASTn + (qt + 1) * 128) / 64;   // causal chunk count
  const float scale = 0.08838834764831845f;        // 1/sqrt(128)

  // one K chunk (64x128) + one V chunk (128x64): 2x1024 chunks of 16B, 8/thread
  auto issue_kv = [&](int buf, int sc2) {
    int s0 = sc2 * 64;
    unsigned lk = ks_u32 + (unsigned)buf * 16384u;
    unsigned lv = vs_u32 + (unsigned)buf * 16384u;
#pragma unroll
    for (int i = 0; i < 4; ++i) {
      int c = tid + i * 256;
      async_ld_b128(lk + ((unsigned)c << 4),
                    Kall + ((size_t)(s0 + (c >> 4)) * KVh + kvh) * Dd + ((c & 15) << 3));
      async_ld_b128(lv + ((unsigned)c << 4),
                    Vt + (size_t)(kvh * Dd + (c >> 3)) * Sall + s0 + ((c & 7) << 3));
    }
  };

  issue_kv(0, 0);
  for (int sc = 0; sc < nch; ++sc) {
    const int s0 = sc * 64;
    if (sc + 1 < nch) {
      issue_kv((sc + 1) & 1, sc + 1);
      wait_async<8>();          // current K+V landed; next chunk in flight
    } else {
      wait_async<0>();
    }
    __syncthreads();
    const bf16* Ksb = &Ks[sc & 1][0];
    const bf16* Vsb = &Vs[sc & 1][0];

    // scores S = Q * K^T : 4 key subtiles x 4 k-steps
    f32x8 sa[4];
#pragma unroll
    for (int sn = 0; sn < 4; ++sn) {
#pragma unroll
      for (int e = 0; e < 8; ++e) sa[sn][e] = 0.0f;
#pragma unroll
      for (int c = 0; c < 4; ++c) {
        bf16x16 kb = ld_frag(Ksb, sn * 16, 128, c * 32, lane);
        sa[sn] = __builtin_amdgcn_wmma_f32_16x16x32_bf16(
            false, qa[c], false, kb, (short)0, sa[sn], false, false);
      }
    }

    // online softmax per query row (row v lives in half-wave, 16 lanes)
#pragma unroll
    for (int v = 0; v < 8; ++v) {
      int m_loc = v + ((lane >> 4) << 3);
      int qg = q0 + m_loc;
      float sv[4];
      float rmax = -1e30f;
#pragma unroll
      for (int sn = 0; sn < 4; ++sn) {
        int key = s0 + sn * 16 + (lane & 15);
        float t = sa[sn][v] * scale;
        if (key > PASTn + qg) t = -1e30f;
        sv[sn] = t;
        rmax = fmaxf(rmax, t);
      }
#pragma unroll
      for (int off = 1; off < 16; off <<= 1)
        rmax = fmaxf(rmax, __shfl_xor(rmax, off, 32));
      float newm = fmaxf(mi[v], rmax);
      float alpha = __expf(mi[v] - newm);
      float rsum = 0.0f;
#pragma unroll
      for (int sn = 0; sn < 4; ++sn) {
        float p = __expf(sv[sn] - newm);
        sv[sn] = p;
        rsum += p;
      }
#pragma unroll
      for (int off = 1; off < 16; off <<= 1)
        rsum += __shfl_xor(rsum, off, 32);
      li[v] = li[v] * alpha + rsum;
      mi[v] = newm;
#pragma unroll
      for (int nt = 0; nt < 8; ++nt)
#pragma unroll
        for (int e = 0; e < 8; ++e) { if (e == v) o[nt][e] *= alpha; }
      // bounce P through LDS to reshape C-layout -> A-fragment layout
#pragma unroll
      for (int sn = 0; sn < 4; ++sn)
        Ps[w * 1024 + m_loc * 64 + sn * 16 + (lane & 15)] = to_bf16(sv[sn]);
    }

    // O += P * V : 2 k-steps of 32 keys, 8 d-subtiles
#pragma unroll
    for (int ks2 = 0; ks2 < 2; ++ks2) {
      bf16x16 pf = ld_frag(Ps + w * 1024, 0, 64, ks2 * 32, lane);
#pragma unroll
      for (int nt = 0; nt < 8; ++nt) {
        bf16x16 vf = ld_frag(Vsb, nt * 16, 64, ks2 * 32, lane);
        o[nt] = __builtin_amdgcn_wmma_f32_16x16x32_bf16(
            false, pf, false, vf, (short)0, o[nt], false, false);
      }
    }
    __syncthreads();
  }

  // finalize: divide by row sums, emit bf16 for the output projection
#pragma unroll
  for (int nt = 0; nt < 8; ++nt)
#pragma unroll
    for (int v = 0; v < 8; ++v) {
      int t = q0 + v + ((lane >> 4) << 3);
      int d = nt * 16 + (lane & 15);
      Ob[(size_t)t * (Hh * Dd) + h * Dd + d] = to_bf16(o[nt][v] / li[v]);
    }
}

// ---------------- host-side orchestration ----------------
extern "C" void kernel_launch(void* const* d_in, const int* in_sizes, int n_in,
                              void* d_out, int out_size, void* d_ws, size_t ws_size,
                              hipStream_t stream) {
  const float* x   = (const float*)d_in[0];
  const float* ve  = (const float*)d_in[1];
  const float* cosp= (const float*)d_in[2];
  const float* sinp= (const float*)d_in[3];
  const float* pk  = (const float*)d_in[4];
  const float* pv  = (const float*)d_in[5];
  const float* Wq  = (const float*)d_in[6];
  const float* Wk  = (const float*)d_in[7];
  const float* Wv  = (const float*)d_in[8];
  const float* Wo  = (const float*)d_in[9];
  const float* Wg  = (const float*)d_in[10];

  float* y_out = (float*)d_out;                                  // [T][C]
  float* pk_out = y_out + (size_t)Tq * Cdim;                     // [S][KV][D]
  float* pv_out = pk_out + (size_t)Sall * KVh * Dd;              // [S][KV][D]

  // workspace carving (256B aligned)
  char* ws = (char*)d_ws;
  size_t off = 0;
  auto carve = [&](size_t bytes) -> void* {
    off = (off + 255) & ~(size_t)255;
    void* p = ws + off;
    off += bytes;
    return p;
  };
  bf16*  xb    = (bf16*)carve((size_t)Tq * Cdim * 2);
  bf16*  WcatT = (bf16*)carve((size_t)NQKV * Cdim * 2);          // [N][K]
  bf16*  WobT  = (bf16*)carve((size_t)Cdim * (Hh * Dd) * 2);     // [N][K]
  float* qkvf  = (float*)carve((size_t)Tq * NQKV * 4);
  bf16*  qb    = (bf16*)carve((size_t)Tq * Hh * Dd * 2);
  bf16*  Kallb = (bf16*)carve((size_t)Sall * KVh * Dd * 2);      // [S][KV][D]
  bf16*  Vtb   = (bf16*)carve((size_t)KVh * Dd * Sall * 2);      // [KV][D][S]
  bf16*  attnb = (bf16*)carve((size_t)Tq * Hh * Dd * 2);
  (void)ws_size; (void)n_in; (void)in_sizes; (void)out_size;

  // 1) casts (weights transposed to [N][K] for async-friendly GEMM B tiles)
  {
    int n = Tq * Cdim;
    cast_f32_bf16<<<(n + 255) / 256, 256, 0, stream>>>(x, xb, n);
  }
  {
    int n = Cdim * (Hh * Dd);
    cast_pack_T<<<(n + 255) / 256, 256, 0, stream>>>(Wq, WcatT, Cdim, Hh * Dd, Cdim, 0);
    int n2 = Cdim * (KVh * Dd);
    cast_pack_T<<<(n2 + 255) / 256, 256, 0, stream>>>(Wk, WcatT, Cdim, KVh * Dd, Cdim, Hh * Dd);
    cast_pack_T<<<(n2 + 255) / 256, 256, 0, stream>>>(Wv, WcatT, Cdim, KVh * Dd, Cdim, Hh * Dd + KVh * Dd);
    int n3 = (Hh * Dd) * Cdim;
    cast_pack_T<<<(n3 + 255) / 256, 256, 0, stream>>>(Wo, WobT, Hh * Dd, Cdim, Hh * Dd, 0);
  }

  // 2) fused QKV projection: [T x C] * [C x 3072]
  {
    dim3 grid(NQKV / 128, Tq / 128);
    gemm_bf16t<<<grid, 256, 0, stream>>>(xb, WcatT, qkvf,
                                         Tq, NQKV, Cdim, Cdim, NQKV);
  }

  // 3) RoPE/RMSNorm/gating + cache writes
  postproc_q<<<Tq * Hh, 128, 0, stream>>>(qkvf, cosp, sinp, qb);
  postproc_kv<<<Tq * KVh, 128, 0, stream>>>(qkvf, cosp, sinp, x, ve, Wg,
                                            pk_out, pv_out, Kallb, Vtb);
  {
    int n = PASTn * KVh * Dd;
    copy_past<<<(n + 255) / 256, 256, 0, stream>>>(pk, pv, pk_out, pv_out,
                                                   Kallb, Vtb, n);
  }

  // 4) flash attention
  {
    dim3 grid(Tq / 128, Hh);
    flash_attn<<<grid, 256, 0, stream>>>(qb, Kallb, Vtb, attnb);
  }

  // 5) output projection: [T x 2048] * [2048 x 2048] -> y (f32)
  {
    dim3 grid(Cdim / 128, Tq / 128);
    gemm_bf16t<<<grid, 256, 0, stream>>>(attnb, WobT, y_out,
                                         Tq, Cdim, Hh * Dd, Hh * Dd, Cdim);
  }
}
